// model_31284541784297
// MI455X (gfx1250) — matrix operand order repaired and checked by reading, not compiled, not run on hardware
//
#include <hip/hip_runtime.h>
#include <hip/hip_fp16.h>

#define TSEQ   2048
#define HID    64
#define VOCAB  50257
#define NPAD   50304          // VOCAB rounded up to multiple of 64
#define NGRPS  (NPAD / 64)    // 786 groups of 4 n-tiles
#define MTILES (TSEQ / 16)    // 128

typedef __attribute__((ext_vector_type(16))) _Float16 v16h;
typedef __attribute__((ext_vector_type(8)))  _Float16 v8h;
typedef __attribute__((ext_vector_type(8)))  float    v8f;

// ---- workspace layout (bytes) ----  (~6.8 MB total)
#define WS_CTX 0          // 64 f32
#define WS_HT  256        // 64 f32
#define WS_CT  512        // 64 f32
#define WS_ARG 1024       // 2048 u64  (ends 17408)
#define WS_A   32768      // 2048*64 f16 = 256 KB (ends 294912)
#define WS_W   294912     // 50304*64 f16 = 6.44 MB

__device__ __forceinline__ float sigmoidf_(float x) {
    return 1.0f / (1.0f + __expf(-x));
}

// =====================================================================
// Encoder LSTM: one workgroup, 256 threads, thread j owns gate row j.
// Weight rows live in registers (fully unrolled). Produces hT, cT, and
// ctx = sum_t h_t (the constant "attention" context -- see analysis).
// =====================================================================
__global__ __launch_bounds__(256) void enc_kernel(
    const int* __restrict__ seq, const float* __restrict__ embed,
    const float* __restrict__ Wih, const float* __restrict__ Whh,
    const float* __restrict__ bih, const float* __restrict__ bhh,
    float* __restrict__ ws_ctx, float* __restrict__ ws_h, float* __restrict__ ws_c)
{
    __shared__ float xs[HID], hs[HID], gates[4 * HID];
    const int j = threadIdx.x;

    float rih[HID], rhh[HID];
#pragma unroll
    for (int k = 0; k < HID; ++k) { rih[k] = Wih[j * HID + k]; rhh[k] = Whh[j * HID + k]; }
    const float bsum = bih[j] + bhh[j];

    float c = 0.f, h = 0.f, ctx = 0.f;
    if (j < HID) hs[j] = 0.f;
    __syncthreads();

    for (int t = 0; t < TSEQ; ++t) {
        if (j < HID) xs[j] = embed[(long long)seq[t] * HID + j];
        __syncthreads();

        float acc = bsum;
#pragma unroll
        for (int k = 0; k < HID; ++k)
            acc = fmaf(rih[k], xs[k], fmaf(rhh[k], hs[k], acc));

        // torch gate order i,f,g,o by row block: sigmoid for i,f,o; tanh for g
        if (j < 2 * HID || j >= 3 * HID) acc = sigmoidf_(acc);
        else                             acc = tanhf(acc);
        gates[j] = acc;
        __syncthreads();

        if (j < HID) {
            float ig = gates[j], fg = gates[HID + j];
            float gg = gates[2 * HID + j], og = gates[3 * HID + j];
            c = fmaf(fg, c, ig * gg);
            h = og * tanhf(c);
            ctx += h;
            hs[j] = h;
        }
        __syncthreads();
    }
    if (j < HID) { ws_ctx[j] = ctx; ws_h[j] = h; ws_c[j] = c; }
}

// =====================================================================
// Decoder LSTM: one workgroup, 256 threads. comb_w (32 KB) in LDS,
// dec weight rows in registers. Writes h2[t] directly as f16 A-matrix.
// =====================================================================
__global__ __launch_bounds__(256) void dec_kernel(
    const int* __restrict__ gold, const float* __restrict__ embed,
    const float* __restrict__ Wih, const float* __restrict__ Whh,
    const float* __restrict__ bih, const float* __restrict__ bhh,
    const float* __restrict__ comb_w, const float* __restrict__ comb_b,
    const float* __restrict__ ws_ctx, const float* __restrict__ ws_h,
    const float* __restrict__ ws_c, _Float16* __restrict__ Af16)
{
    __shared__ float comb[HID * 2 * HID];   // 64 x 128 = 32 KB
    __shared__ float xs[HID], hs[HID], ahs[HID], ctxs[HID], gates[4 * HID];
    const int j = threadIdx.x;

    for (int i = j; i < HID * 2 * HID; i += 256) comb[i] = comb_w[i];

    float rih[HID], rhh[HID];
#pragma unroll
    for (int k = 0; k < HID; ++k) { rih[k] = Wih[j * HID + k]; rhh[k] = Whh[j * HID + k]; }
    const float bsum = bih[j] + bhh[j];

    float c = 0.f;
    if (j < HID) { hs[j] = ws_h[j]; ctxs[j] = ws_ctx[j]; c = ws_c[j]; }
    __syncthreads();

    for (int t = 0; t < TSEQ; ++t) {
        // teacher-forced input: relu(embed[gold[t-1]]), zeros at t=0
        if (j < HID)
            xs[j] = (t == 0) ? 0.f
                             : fmaxf(0.f, embed[(long long)gold[t - 1] * HID + j]);
        __syncthreads();

        // ah = comb_w @ [h ; ctx] + comb_b
        if (j < HID) {
            float a = comb_b[j];
#pragma unroll
            for (int k = 0; k < HID; ++k) a = fmaf(comb[j * 2 * HID + k], hs[k], a);
#pragma unroll
            for (int k = 0; k < HID; ++k) a = fmaf(comb[j * 2 * HID + HID + k], ctxs[k], a);
            ahs[j] = a;
        }
        __syncthreads();

        float acc = bsum;
#pragma unroll
        for (int k = 0; k < HID; ++k)
            acc = fmaf(rih[k], xs[k], fmaf(rhh[k], ahs[k], acc));
        if (j < 2 * HID || j >= 3 * HID) acc = sigmoidf_(acc);
        else                             acc = tanhf(acc);
        gates[j] = acc;
        __syncthreads();

        if (j < HID) {
            float ig = gates[j], fg = gates[HID + j];
            float gg = gates[2 * HID + j], og = gates[3 * HID + j];
            c = fmaf(fg, c, ig * gg);
            float h2 = og * tanhf(c);
            hs[j] = h2;
            Af16[t * HID + j] = (_Float16)h2;
        }
        __syncthreads();
    }
}

// =====================================================================
// Prep: f32->f16 zero-padded copy of out_w (B matrix), init argmax buf
// =====================================================================
__global__ void prep_kernel(const float* __restrict__ out_w,
                            _Float16* __restrict__ Bf16,
                            unsigned long long* __restrict__ argbuf)
{
    const int i = blockIdx.x * blockDim.x + threadIdx.x;
    if (i < NPAD * HID)
        Bf16[i] = (i < VOCAB * HID) ? (_Float16)out_w[i] : (_Float16)0.f;
    if (i < TSEQ) argbuf[i] = 0ull;
}

// =====================================================================
// Output GEMM: preds[2048, 50257] = A[2048,64] @ out_w^T + out_b
// One wave per 16(M) x 64(N) strip = 4 n-tiles: A fragments loaded once,
// 4 independent WMMA accumulator chains (8 x v_wmma_f32_16x16x32_f16).
// Fused row-argmax: 4 candidates reduced per-lane first, then one
// 16-lane shuffle reduction + global_atomic_max_u64 per row.
// =====================================================================
__global__ __launch_bounds__(256) void gemm_kernel(
    const _Float16* __restrict__ A, const _Float16* __restrict__ B,
    const float* __restrict__ out_b, float* __restrict__ preds,
    unsigned long long* __restrict__ argbuf)
{
    const int wave  = threadIdx.x >> 5;
    const int lane  = threadIdx.x & 31;
    const int strip = blockIdx.x * 8 + wave;     // 128*786 strips total
    const int mTile = strip / NGRPS;
    const int nBase = (strip % NGRPS) * 64;
    const int half  = lane >> 4;
    const int ln    = lane & 15;

    // ---- A fragments (16x32 f16 each): lane row = M, chunks at K=half*8, 16+half*8
    const _Float16* ap = A + (mTile * 16 + ln) * HID;
    v8h a0lo = *(const v8h*)(ap + half * 8);
    v8h a0hi = *(const v8h*)(ap + 16 + half * 8);
    v8h a1lo = *(const v8h*)(ap + 32 + half * 8);
    v8h a1hi = *(const v8h*)(ap + 48 + half * 8);
    v16h a0, a1;
#pragma unroll
    for (int i = 0; i < 8; ++i) {
        a0[i] = a0lo[i]; a0[8 + i] = a0hi[i];
        a1[i] = a1lo[i]; a1[8 + i] = a1hi[i];
    }

    // ---- 4 independent accumulator chains over the 64-wide N strip
    v8f cc[4];
#pragma unroll
    for (int g = 0; g < 4; ++g) {
        // B fragments (32x16 f16): lane col = N, chunks at K=half*8, 16+half*8
        const _Float16* bp = B + (nBase + g * 16 + ln) * HID;
        v8h b0lo = *(const v8h*)(bp + half * 8);
        v8h b0hi = *(const v8h*)(bp + 16 + half * 8);
        v8h b1lo = *(const v8h*)(bp + 32 + half * 8);
        v8h b1hi = *(const v8h*)(bp + 48 + half * 8);
        v16h b0, b1;
#pragma unroll
        for (int i = 0; i < 8; ++i) {
            b0[i] = b0lo[i]; b0[8 + i] = b0hi[i];
            b1[i] = b1lo[i]; b1[8 + i] = b1hi[i];
        }
        v8f acc = {};
        acc = __builtin_amdgcn_wmma_f32_16x16x32_f16(false, a0, false, b0,
                                                     (short)0, acc, false, false);
        acc = __builtin_amdgcn_wmma_f32_16x16x32_f16(false, a1, false, b1,
                                                     (short)0, acc, false, false);
        cc[g] = acc;
    }

    // per-g column index / validity / bias
    int   nCol[4];
    bool  nVal[4];
    float bias[4];
#pragma unroll
    for (int g = 0; g < 4; ++g) {
        nCol[g] = nBase + g * 16 + ln;
        nVal[g] = (nCol[g] < VOCAB);
        bias[g] = nVal[g] ? out_b[nCol[g]] : 0.f;
    }

#pragma unroll
    for (int r = 0; r < 8; ++r) {
        const int m = mTile * 16 + half * 8 + r;
        unsigned long long pk = 0ull;
#pragma unroll
        for (int g = 0; g < 4; ++g) {
            const float v = cc[g][r] + bias[g];
            if (nVal[g]) preds[(long long)m * VOCAB + nCol[g]] = v;

            // order-preserving float key; ~n in low bits => ties -> lowest index
            unsigned int bits = __float_as_uint(v);
            unsigned int key  = (bits & 0x80000000u) ? ~bits : (bits | 0x80000000u);
            unsigned long long cand =
                ((unsigned long long)key << 32) | (unsigned int)(~nCol[g]);
            if (!nVal[g]) cand = 0ull;
            pk = (cand > pk) ? cand : pk;         // per-lane reduce over 4 cols
        }
#pragma unroll
        for (int off = 8; off >= 1; off >>= 1) {  // 16-lane reduce within half
            unsigned long long o = __shfl_xor(pk, off, 16);
            pk = (o > pk) ? o : pk;
        }
        if (ln == 0) atomicMax(&argbuf[m], pk);
    }
}

// =====================================================================
// Finalize: decode packed argmax -> float index in d_out tail
// =====================================================================
__global__ void fin_kernel(const unsigned long long* __restrict__ argbuf,
                           float* __restrict__ out_idx)
{
    const int t = blockIdx.x * blockDim.x + threadIdx.x;
    if (t < TSEQ) {
        unsigned int low = (unsigned int)(argbuf[t] & 0xFFFFFFFFull);
        out_idx[t] = (float)(~low);
    }
}

extern "C" void kernel_launch(void* const* d_in, const int* in_sizes, int n_in,
                              void* d_out, int out_size, void* d_ws, size_t ws_size,
                              hipStream_t stream)
{
    (void)in_sizes; (void)n_in; (void)out_size; (void)ws_size;

    const int*   input_seq = (const int*)  d_in[0];
    const int*   gold_seq  = (const int*)  d_in[1];
    const float* embed     = (const float*)d_in[2];
    const float* enc_Wih   = (const float*)d_in[3];
    const float* enc_Whh   = (const float*)d_in[4];
    const float* enc_bih   = (const float*)d_in[5];
    const float* enc_bhh   = (const float*)d_in[6];
    const float* dec_Wih   = (const float*)d_in[7];
    const float* dec_Whh   = (const float*)d_in[8];
    const float* dec_bih   = (const float*)d_in[9];
    const float* dec_bhh   = (const float*)d_in[10];
    // d_in[11..16] = U_w,U_b,W_w,W_b,V_w,V_b : dead code (softmax over axis=1
    // of a [T,1] tensor is identically 1 => ctx is a constant column-sum)
    const float* comb_w    = (const float*)d_in[17];
    const float* comb_b    = (const float*)d_in[18];
    const float* out_w     = (const float*)d_in[19];
    const float* out_b     = (const float*)d_in[20];

    char* ws = (char*)d_ws;
    float*              ws_ctx = (float*)(ws + WS_CTX);
    float*              ws_h   = (float*)(ws + WS_HT);
    float*              ws_c   = (float*)(ws + WS_CT);
    unsigned long long* argbuf = (unsigned long long*)(ws + WS_ARG);
    _Float16*           Af16   = (_Float16*)(ws + WS_A);
    _Float16*           Bf16   = (_Float16*)(ws + WS_W);

    float* preds   = (float*)d_out;
    float* out_idx = preds + (long long)TSEQ * VOCAB;

    // prep (B f16 conversion + argmax init): 50304*64 elements
    const int prep_n = NPAD * HID;
    prep_kernel<<<(prep_n + 255) / 256, 256, 0, stream>>>(out_w, Bf16, argbuf);

    // sequential encoder scan
    enc_kernel<<<1, 256, 0, stream>>>(input_seq, embed,
                                      enc_Wih, enc_Whh, enc_bih, enc_bhh,
                                      ws_ctx, ws_h, ws_c);

    // sequential decoder scan -> A matrix (f16)
    dec_kernel<<<1, 256, 0, stream>>>(gold_seq, embed,
                                      dec_Wih, dec_Whh, dec_bih, dec_bhh,
                                      comb_w, comb_b,
                                      ws_ctx, ws_h, ws_c, Af16);

    // output GEMM + fused argmax: 128 * 786 strips, 8 waves/block
    gemm_kernel<<<(MTILES * NGRPS) / 8, 256, 0, stream>>>(Af16, Bf16, out_b,
                                                          preds, argbuf);

    // decode argmax indices
    fin_kernel<<<(TSEQ + 255) / 256, 256, 0, stream>>>(argbuf, out_idx);
}
